// LstmEncDeltaAllHistStacked_48670569398991
// MI455X (gfx1250) — matrix operand order repaired and checked by reading, not compiled, not run on hardware
//
#include <hip/hip_runtime.h>
#include <hip/hip_bf16.h>

// ---------------------------------------------------------------------------
// Problem constants (from reference setup_inputs)
// ---------------------------------------------------------------------------
#define NPD   256   // number of persons
#define SEQL  8     // sequence length
#define DIN   2     // raw input dim
#define HID64 64    // hidden size of node/edge/seq LSTMs
#define G256  256   // 4*HID64
#define EMB   32    // decoder hidden
#define G128  128   // 4*EMB

typedef __attribute__((ext_vector_type(16))) _Float16 v16h;
typedef __attribute__((ext_vector_type(8)))  float    v8f;

// ---------------------------------------------------------------------------
// WMMA fragment loaders (wave32; M/N and C/D per CDNA5 ISA 7.12.2, A/B k order contiguous)
//
// A (16x32 f16, MxK): lane L: M = L%16.
//   halves 0..15 -> K = 16*(L/16) + h (same order as B)
// B (32x16 f16, KxN): lane L: N = L%16, half h -> K = 16*(L/16) + h.
//   With Whh stored row-major [4H][H] (PyTorch layout), B[k][n] = Whh[n][k]
//   is contiguous in h.
// C/D (v8f): reg r, lane L -> (M = r + 8*(L/16), N = L%16).
// ---------------------------------------------------------------------------
__device__ __forceinline__ v16h load_a_frag(const _Float16* hbuf, int ld,
                                            int k0, int lane) {
  const int m    = lane & 15;
  const int base = k0 + ((lane >> 4) << 4);   // +16 for upper half-wave
  const _Float16* row = hbuf + m * ld + base;
  v16h a;
#pragma unroll
  for (int t = 0; t < 8; ++t) {
    a[t]     = row[t];        // K = base + t
    a[t + 8] = row[t + 8];    // K = base + 8 + t
  }
  return a;
}

__device__ __forceinline__ v16h load_b_frag(const _Float16* w, int ldk,
                                            int n, int k0, int lane) {
  const _Float16* row = w + n * ldk + k0 + ((lane >> 4) << 4);
  v16h b;
#pragma unroll
  for (int t = 0; t < 16; ++t) b[t] = row[t];
  return b;
}

// Branchless fast nonlinearities: native v_exp_f32 + v_rcp_f32, no EXEC
// branches inside the recurrence loop (co-executes with XDL WMMAs).
__device__ __forceinline__ float sigm(float x) {
  return __builtin_amdgcn_rcpf(1.0f + __expf(-x));
}
__device__ __forceinline__ float tanh_fast(float x) {
  return 2.0f * __builtin_amdgcn_rcpf(1.0f + __expf(-2.0f * x)) - 1.0f;
}

// ---------------------------------------------------------------------------
// f32 -> f16 weight conversion
// ---------------------------------------------------------------------------
__global__ void cvt_f16_kernel(const float* __restrict__ src,
                               _Float16* __restrict__ dst, int n) {
  int i = blockIdx.x * blockDim.x + threadIdx.x;
  if (i < n) dst[i] = (_Float16)src[i];
}

// ---------------------------------------------------------------------------
// Generic input projection: xg[t][b][n] = (bih[n]+bhh[n]) +
//     sum_k src1[b][t][k] * Wih[n][k] + sum_k src2[b][t][k] * Wih[n][in1+k]
// Layouts: src [B][T][In] row-major, xg [T][B][G] row-major.
// src2 == nullptr  -> single input;  bih == nullptr -> no bias (edge P table).
// ---------------------------------------------------------------------------
__global__ void proj_kernel(const float* __restrict__ src1, int in1,
                            const float* __restrict__ src2, int in2,
                            const float* __restrict__ Wih,
                            const float* __restrict__ bih,
                            const float* __restrict__ bhh,
                            float* __restrict__ xg, int T, int B, int G) {
  int idx = blockIdx.x * blockDim.x + threadIdx.x;
  int total = T * B * G;
  if (idx >= total) return;
  int n = idx % G;
  int b = (idx / G) % B;
  int t = idx / (G * B);
  int IN = in1 + in2;
  float acc = bih ? (bih[n] + bhh[n]) : 0.0f;
  const float* wr = Wih + (size_t)n * IN;
  const float* x1 = src1 + ((size_t)b * T + t) * in1;
  for (int k = 0; k < in1; ++k) acc += x1[k] * wr[k];
  if (src2) {
    const float* x2 = src2 + ((size_t)b * T + t) * in2;
    for (int k = 0; k < in2; ++k) acc += x2[k] * wr[in1 + k];
  }
  xg[idx] = acc;   // idx == (t*B + b)*G + n
}

// ---------------------------------------------------------------------------
// Edge LSTM: 2048 rows (i,s), scanned over j = 0..255 neighbors.
// Block = 16 rows sharing the same s (so P[s][j] is block-uniform per step).
// Gin[m][n] = bias[n] + P[s][j][n] - P[s][i_m][n]  (delta trick).
// Keeps only final hidden -> dist_hist[i][s][:].
// ---------------------------------------------------------------------------
__global__ __launch_bounds__(128) void edge_lstm_kernel(
    const float* __restrict__ P,        // [SEQ][NP][256]
    const _Float16* __restrict__ WhhH,  // [256][64] f16 (PyTorch layout)
    const float* __restrict__ bih, const float* __restrict__ bhh,
    float* __restrict__ dist_hist) {    // [NP][SEQ][64]
  __shared__ __align__(16) _Float16 sWhh[G256 * HID64];  // 32 KB
  __shared__ __align__(16) float    sPi[16 * G256];      // 16 KB
  __shared__ __align__(16) _Float16 sH[16 * HID64];      // 2 KB

  const int tid  = threadIdx.x;
  const int lane = tid & 31;
  const int wave = tid >> 5;              // 0..3
  const int ig   = blockIdx.x >> 3;       // person-group 0..15
  const int s    = blockIdx.x & 7;
  const int i0   = ig * 16;

  // Stage Whh (f16) into LDS as 16B chunks.
  {
    const uint4* src = (const uint4*)WhhH;
    uint4* dst = (uint4*)sWhh;
    for (int idx = tid; idx < (G256 * HID64 * 2) / 16; idx += 128)
      dst[idx] = src[idx];
  }
  // sPi[m][n] = P[s][i0+m][n] - (bih+bhh)[n]  =>  Gin = P[s][j][n] - sPi
  for (int idx = tid; idx < 16 * G256; idx += 128) {
    int m = idx >> 8, n = idx & 255;
    sPi[idx] = P[((size_t)s * NPD + (i0 + m)) * G256 + n] - (bih[n] + bhh[n]);
  }
  for (int idx = tid; idx < 16 * HID64; idx += 128) sH[idx] = (_Float16)0.0f;
  __syncthreads();

  float c[8];
#pragma unroll
  for (int r = 0; r < 8; ++r) c[r] = 0.0f;

  const int col   = lane & 15;
  const int half  = lane >> 4;
  const int j_hid = wave * 16 + col;      // hidden column owned by this lane
  float hv[8];

  for (int j = 0; j < NPD; ++j) {
    const float* Pj = P + ((size_t)s * NPD + j) * G256;
    v8f acc[4];
#pragma unroll
    for (int g = 0; g < 4; ++g) {
      const int n = g * HID64 + j_hid;
      const float pj = Pj[n];
      v8f cf;
#pragma unroll
      for (int r = 0; r < 8; ++r) cf[r] = pj - sPi[(r + half * 8) * G256 + n];
#pragma unroll
      for (int k0 = 0; k0 < HID64; k0 += 32) {
        v16h a = load_a_frag(sH, HID64, k0, lane);
        v16h b = load_b_frag(sWhh, HID64, n, k0, lane);
        cf = __builtin_amdgcn_wmma_f32_16x16x32_f16(
            false, a, false, b, (short)0, cf, false, false);
      }
      acc[g] = cf;
    }
    __syncthreads();   // all reads of sH complete
#pragma unroll
    for (int r = 0; r < 8; ++r) {
      float iv = sigm(acc[0][r]);
      float fv = sigm(acc[1][r]);
      float gv = tanh_fast(acc[2][r]);
      float ov = sigm(acc[3][r]);
      c[r]  = fv * c[r] + iv * gv;
      hv[r] = ov * tanh_fast(c[r]);
      sH[(r + half * 8) * HID64 + j_hid] = (_Float16)hv[r];
    }
    __syncthreads();   // new h visible to all waves
  }

  // Final hidden -> dist_hist[i][s][:]
#pragma unroll
  for (int r = 0; r < 8; ++r) {
    int m = r + half * 8;
    dist_hist[(((size_t)(i0 + m)) * SEQL + s) * HID64 + j_hid] = hv[r];
  }
}

// ---------------------------------------------------------------------------
// Generic small LSTM (node / seq / dec): 16 batch rows per block,
// HID/16 waves, collects every timestep. xg precomputed as [T][B][4H].
// ---------------------------------------------------------------------------
template <int HID>
__global__ __launch_bounds__(128) void lstm_kernel(
    const float* __restrict__ xg,       // [T][B][4*HID]
    const _Float16* __restrict__ WhhH,  // [4*HID][HID] f16
    float* __restrict__ out,            // [B][T][HID]
    int T, int B) {
  constexpr int G = 4 * HID;
  __shared__ __align__(16) _Float16 sWhh[G * HID];
  __shared__ __align__(16) _Float16 sH[16 * HID];

  const int tid  = threadIdx.x;
  const int lane = tid & 31;
  const int wave = tid >> 5;
  const int nthr = (HID / 16) * 32;
  const int b0   = blockIdx.x * 16;

  {
    const uint4* src = (const uint4*)WhhH;
    uint4* dst = (uint4*)sWhh;
    for (int idx = tid; idx < (G * HID * 2) / 16; idx += nthr) dst[idx] = src[idx];
  }
  for (int idx = tid; idx < 16 * HID; idx += nthr) sH[idx] = (_Float16)0.0f;
  __syncthreads();

  float c[8];
#pragma unroll
  for (int r = 0; r < 8; ++r) c[r] = 0.0f;

  const int col   = lane & 15;
  const int half  = lane >> 4;
  const int j_hid = wave * 16 + col;

  for (int t = 0; t < T; ++t) {
    const float* xgt = xg + ((size_t)t * B + b0) * G;
    v8f acc[4];
#pragma unroll
    for (int g = 0; g < 4; ++g) {
      const int n = g * HID + j_hid;
      v8f cf;
#pragma unroll
      for (int r = 0; r < 8; ++r) cf[r] = xgt[(size_t)(r + half * 8) * G + n];
#pragma unroll
      for (int k0 = 0; k0 < HID; k0 += 32) {
        v16h a = load_a_frag(sH, HID, k0, lane);
        v16h b = load_b_frag(sWhh, HID, n, k0, lane);
        cf = __builtin_amdgcn_wmma_f32_16x16x32_f16(
            false, a, false, b, (short)0, cf, false, false);
      }
      acc[g] = cf;
    }
    __syncthreads();
#pragma unroll
    for (int r = 0; r < 8; ++r) {
      float iv = sigm(acc[0][r]);
      float fv = sigm(acc[1][r]);
      float gv = tanh_fast(acc[2][r]);
      float ov = sigm(acc[3][r]);
      c[r] = fv * c[r] + iv * gv;
      float hval = ov * tanh_fast(c[r]);
      int m = r + half * 8;
      sH[m * HID + j_hid] = (_Float16)hval;
      out[((size_t)(b0 + m) * T + t) * HID + j_hid] = hval;
    }
    __syncthreads();
  }
}

// ---------------------------------------------------------------------------
// Pose head: tag[i][d] = scene[i][SEQ-1][d] + pose_b[d]
//                        + sum_k decoded_flat[i][k] * pose_W[d][k]
// ---------------------------------------------------------------------------
__global__ void pose_kernel(const float* __restrict__ scene,
                            const float* __restrict__ decoded,  // [NP][SEQ][EMB]
                            const float* __restrict__ poseW,    // [2][SEQ*EMB]
                            const float* __restrict__ poseb,
                            float* __restrict__ out) {          // [NP][1][2]
  int idx = blockIdx.x * blockDim.x + threadIdx.x;
  if (idx >= NPD * 2) return;
  int i = idx >> 1, d = idx & 1;
  constexpr int K = SEQL * EMB;  // 256
  float acc = scene[((size_t)i * SEQL + (SEQL - 1)) * DIN + d] + poseb[d];
  const float* w  = poseW + (size_t)d * K;
  const float* dr = decoded + (size_t)i * K;
  for (int k = 0; k < K; ++k) acc += dr[k] * w[k];
  out[idx] = acc;
}

// ---------------------------------------------------------------------------
// Orchestration
// ---------------------------------------------------------------------------
extern "C" void kernel_launch(void* const* d_in, const int* in_sizes, int n_in,
                              void* d_out, int out_size, void* d_ws, size_t ws_size,
                              hipStream_t stream) {
  (void)in_sizes; (void)n_in; (void)out_size; (void)ws_size;

  const float* scene    = (const float*)d_in[0];
  const float* node_Wih = (const float*)d_in[1];
  const float* node_Whh = (const float*)d_in[2];
  const float* node_bih = (const float*)d_in[3];
  const float* node_bhh = (const float*)d_in[4];
  const float* edge_Wih = (const float*)d_in[5];
  const float* edge_Whh = (const float*)d_in[6];
  const float* edge_bih = (const float*)d_in[7];
  const float* edge_bhh = (const float*)d_in[8];
  const float* seq_Wih  = (const float*)d_in[9];
  const float* seq_Whh  = (const float*)d_in[10];
  const float* seq_bih  = (const float*)d_in[11];
  const float* seq_bhh  = (const float*)d_in[12];
  const float* dec_Wih  = (const float*)d_in[13];
  const float* dec_Whh  = (const float*)d_in[14];
  const float* dec_bih  = (const float*)d_in[15];
  const float* dec_bhh  = (const float*)d_in[16];
  const float* pose_W   = (const float*)d_in[17];
  const float* pose_b   = (const float*)d_in[18];
  float* out = (float*)d_out;

  // Workspace carve-out (256B aligned)
  char* ws = (char*)d_ws;
  size_t off = 0;
  auto carve = [&](size_t bytes) -> void* {
    void* p = ws + off;
    off = (off + bytes + 255) & ~(size_t)255;
    return p;
  };
  _Float16* whhN = (_Float16*)carve(G256 * HID64 * sizeof(_Float16));
  _Float16* whhE = (_Float16*)carve(G256 * HID64 * sizeof(_Float16));
  _Float16* whhS = (_Float16*)carve(G256 * HID64 * sizeof(_Float16));
  _Float16* whhD = (_Float16*)carve(G128 * EMB * sizeof(_Float16));
  float* P         = (float*)carve((size_t)SEQL * NPD * G256 * 4);  // 2 MB
  float* xg_node   = (float*)carve((size_t)SEQL * NPD * G256 * 4);  // 2 MB
  float* xg_seq    = (float*)carve((size_t)SEQL * NPD * G256 * 4);  // 2 MB
  float* xg_dec    = (float*)carve((size_t)SEQL * NPD * G128 * 4);  // 1 MB
  float* lstm_out  = (float*)carve((size_t)NPD * SEQL * HID64 * 4);
  float* dist_hist = (float*)carve((size_t)NPD * SEQL * HID64 * 4);
  float* full_dist = (float*)carve((size_t)NPD * SEQL * HID64 * 4);
  float* decoded   = (float*)carve((size_t)NPD * SEQL * EMB * 4);

  // 1) weight f16 conversion
  cvt_f16_kernel<<<(G256 * HID64 + 255) / 256, 256, 0, stream>>>(node_Whh, whhN, G256 * HID64);
  cvt_f16_kernel<<<(G256 * HID64 + 255) / 256, 256, 0, stream>>>(edge_Whh, whhE, G256 * HID64);
  cvt_f16_kernel<<<(G256 * HID64 + 255) / 256, 256, 0, stream>>>(seq_Whh,  whhS, G256 * HID64);
  cvt_f16_kernel<<<(G128 * EMB  + 255) / 256, 256, 0, stream>>>(dec_Whh,  whhD, G128 * EMB);

  // 2) input projections that don't depend on LSTM outputs
  {
    int total = SEQL * NPD * G256;
    proj_kernel<<<(total + 255) / 256, 256, 0, stream>>>(
        scene, DIN, nullptr, 0, node_Wih, node_bih, node_bhh,
        xg_node, SEQL, NPD, G256);
    proj_kernel<<<(total + 255) / 256, 256, 0, stream>>>(
        scene, DIN, nullptr, 0, edge_Wih, nullptr, nullptr,
        P, SEQL, NPD, G256);                       // P[s][j] = Wih_e @ scene[j,s]
  }

  // 3) node LSTM (collect) and edge LSTM (final hidden only)
  lstm_kernel<HID64><<<NPD / 16, 128, 0, stream>>>(xg_node, whhN, lstm_out, SEQL, NPD);
  edge_lstm_kernel<<<128, 128, 0, stream>>>(P, whhE, edge_bih, edge_bhh, dist_hist);

  // 4) seq LSTM over dist_hist
  {
    int total = SEQL * NPD * G256;
    proj_kernel<<<(total + 255) / 256, 256, 0, stream>>>(
        dist_hist, HID64, nullptr, 0, seq_Wih, seq_bih, seq_bhh,
        xg_seq, SEQL, NPD, G256);
  }
  lstm_kernel<HID64><<<NPD / 16, 128, 0, stream>>>(xg_seq, whhS, full_dist, SEQL, NPD);

  // 5) decoder LSTM over concat(lstm_out, full_dist)
  {
    int total = SEQL * NPD * G128;
    proj_kernel<<<(total + 255) / 256, 256, 0, stream>>>(
        lstm_out, HID64, full_dist, HID64, dec_Wih, dec_bih, dec_bhh,
        xg_dec, SEQL, NPD, G128);
  }
  lstm_kernel<EMB><<<NPD / 16, 64, 0, stream>>>(xg_dec, whhD, decoded, SEQL, NPD);

  // 6) pose head
  pose_kernel<<<2, 256, 0, stream>>>(scene, decoded, pose_W, pose_b, out);
}
